// ForwardExecutor_33062658245245
// MI455X (gfx1250) — matrix operand order repaired and checked by reading, not compiled, not run on hardware
//
#include <hip/hip_runtime.h>

typedef __attribute__((ext_vector_type(16))) __bf16 v16bf;
typedef __attribute__((ext_vector_type(8)))  float  v8f;

#define L_LAYERS 8
#define K_NODES  8
#define INDEG    4
#define DD       128
#define BB       4096
#define NN       (1 + L_LAYERS * K_NODES)   // 65
#define ROWS     16
#define STRIDE   136                        // 128 + 8 bf16 pad -> 272B rows, bank-conflict free
#define SLICE    (ROWS * STRIDE)            // 2176 bf16 per node tile
#define NODEOUT_ELEMS (NN * SLICE)          // 141440 bf16 = 282,880 B
#define STAGE_ELEMS   (K_NODES * SLICE)     // 17408  bf16 =  34,816 B
#define SMEM_BYTES    ((NODEOUT_ELEMS + STAGE_ELEMS) * 2)  // 317,696 B < 320 KB

// ---------------------------------------------------------------------------
// Compile-time reproduction of np.random.default_rng(0).integers(...) so the
// static DAG edge list is baked in as constants (SeedSequence + PCG64 + Lemire).
// ---------------------------------------------------------------------------
namespace rg {
typedef unsigned int u32; typedef unsigned long long u64;
typedef unsigned __int128 u128;
constexpr u32 XSHIFT = 16;
constexpr u32 INIT_A = 0x43b0d7e5u, MULT_A = 0x931e8875u;
constexpr u32 INIT_B = 0x8b51f9ddu, MULT_B = 0x58f38dedu;
constexpr u32 MIX_L  = 0xca01f9ddu, MIX_R  = 0x4973f715u;

constexpr u32 hashmix(u32 v, u32& hc) { v ^= hc; hc *= MULT_A; v *= hc; v ^= v >> XSHIFT; return v; }
constexpr u32 mix2(u32 x, u32 y) { u32 r = (x * MIX_L) ^ (y * MIX_R); r ^= r >> XSHIFT; return r; }

constexpr u128 PCG_MULT = (((u128)0x2360ed051fc65da4ULL) << 64) | (u128)0x4385df649fce6645ULL;
struct Pcg { u128 state; u128 inc; bool has32; u32 buf; };
constexpr u64 rotr64(u64 v, unsigned r) { r &= 63u; return r ? (v >> r) | (v << (64u - r)) : v; }
constexpr u64 next64(Pcg& g) {
  g.state = g.state * PCG_MULT + g.inc;
  u64 hi = (u64)(g.state >> 64), lo = (u64)g.state;
  return rotr64(hi ^ lo, (unsigned)(g.state >> 122));
}
constexpr u32 next32(Pcg& g) {
  if (g.has32) { g.has32 = false; return g.buf; }
  u64 n = next64(g); g.has32 = true; g.buf = (u32)(n >> 32); return (u32)n;
}
struct Srcs { int v[L_LAYERS][K_NODES][INDEG]; };
constexpr Srcs build() {
  // SeedSequence(0): entropy = [0], pool size 4
  u32 hc = INIT_A; u32 pool[4] = {0, 0, 0, 0};
  for (int i = 0; i < 4; i++) pool[i] = hashmix(0u, hc);
  for (int s = 0; s < 4; s++)
    for (int d = 0; d < 4; d++)
      if (s != d) pool[d] = mix2(pool[d], hashmix(pool[s], hc));
  // generate_state(4, uint64) as 8 x uint32
  u32 hb = INIT_B; u32 st[8] = {};
  for (int i = 0; i < 8; i++) { u32 v = pool[i & 3]; v ^= hb; hb *= MULT_B; v *= hb; v ^= v >> XSHIFT; st[i] = v; }
  u64 w0 = (u64)st[0] | ((u64)st[1] << 32), w1 = (u64)st[2] | ((u64)st[3] << 32);
  u64 w2 = (u64)st[4] | ((u64)st[5] << 32), w3 = (u64)st[6] | ((u64)st[7] << 32);
  // PCG64 seeding: initstate = (val0<<64)|val1 ; initseq = (val2<<64)|val3
  Pcg g{}; g.state = 0; g.inc = ((((u128)w2 << 64) | w3) << 1) | (u128)1u; g.has32 = false; g.buf = 0;
  g.state = g.state * PCG_MULT + g.inc;
  g.state += (((u128)w0) << 64) | w1;
  g.state = g.state * PCG_MULT + g.inc;
  Srcs out{};
  for (int k = 0; k < K_NODES; k++) for (int j = 0; j < INDEG; j++) out.v[0][k][j] = 0; // rng==0: no draws
  for (int l = 1; l < L_LAYERS; l++) {
    u32 rng = (u32)(K_NODES * l);            // high = 1+8l, rng = high-1
    u32 rex = rng + 1u;
    u32 thresh = (0xFFFFFFFFu - rng) % rex;  // Lemire threshold
    for (int k = 0; k < K_NODES; k++)
      for (int j = 0; j < INDEG; j++) {
        u64 m = (u64)next32(g) * (u64)rex;
        u32 lo = (u32)m;
        if (lo < rex) { while (lo < thresh) { m = (u64)next32(g) * (u64)rex; lo = (u32)m; } }
        out.v[l][k][j] = (int)(m >> 32);
      }
  }
  return out;
}
} // namespace rg

__constant__ rg::Srcs g_srcs = rg::build();

// ---------------------------------------------------------------------------
#if defined(__has_builtin)
#if __has_builtin(__builtin_amdgcn_cvt_pk_bf16_f32)
#define HAVE_CVT_PK_BF16 1
#endif
#endif

// pack two f32 -> packed bf16x2 with RNE, using the VOP3P mix-conversion ops
// (V_FMA_MIXLO_BF16 / V_FMA_MIXHI_BF16, §15.10 opcodes 62/63): bf16(x*1.0+0).
__device__ __forceinline__ unsigned pack2(float a, float b) {
#ifdef HAVE_CVT_PK_BF16
  return __builtin_bit_cast(unsigned, __builtin_amdgcn_cvt_pk_bf16_f32(a, b));
#else
  unsigned r;
  asm("v_fma_mixlo_bf16 %0, %1, 1.0, 0\n\t"
      "v_fma_mixhi_bf16 %0, %2, 1.0, 0"
      : "=&v"(r) : "v"(a), "v"(b));
  return r;
#endif
}
// expand a packed bf16 pair to two f32 (1 VALU op each)
__device__ __forceinline__ float bflo(unsigned u) { return __builtin_bit_cast(float, u << 16); }
__device__ __forceinline__ float bfhi(unsigned u) { return __builtin_bit_cast(float, u & 0xffff0000u); }
// packed bf16 add (V_PK_ADD_BF16, VOP3P op 35)
__device__ __forceinline__ unsigned pk_add_bf16(unsigned a, unsigned b) {
  unsigned r;
  asm("v_pk_add_bf16 %0, %1, %2" : "=v"(r) : "v"(a), "v"(b));
  return r;
}

union ABReg { v16bf v; uint4 q[2]; };

// A operand (16x32 bf16): lane m<16 holds K = h*8..h*8+7 and 16+h*8..16+h*8+7
__device__ __forceinline__ void load_A(const unsigned short* aStage, int lane, ABReg A[4]) {
  const int m = lane & 15, half = lane >> 4;
  const unsigned short* ap = aStage + m * STRIDE + half * 8;
#pragma unroll
  for (int c = 0; c < 4; c++) {
    A[c].q[0] = *(const uint4*)(ap + c * 32);
    A[c].q[1] = *(const uint4*)(ap + c * 32 + 16);
  }
}

// one 16x16 output tile: C += sum_c A_c * B_c ; then bias+relu -> bf16 LDS slot
__device__ __forceinline__ void mm_tile(const ABReg A[4], int node, int tile, int lane,
                                        const unsigned short* __restrict__ Wb,
                                        const float* __restrict__ bias,
                                        unsigned short* dstSlot) {
  const int n15 = lane & 15, half = lane >> 4;
  const int e = tile * 16 + n15;
  // B operand (32x16 bf16): lane n holds K = half*8..half*8+7 and 16+half*8..16+half*8+7 for column e
  const unsigned short* bp = Wb + ((size_t)node * DD + e) * DD + half * 8;
  ABReg Bm[4];
#pragma unroll
  for (int c = 0; c < 4; c++) {
    Bm[c].q[0] = *(const uint4*)(bp + c * 32);
    Bm[c].q[1] = *(const uint4*)(bp + c * 32 + 16);
  }
  v8f acc = {};
#pragma unroll
  for (int c = 0; c < 4; c++)
    acc = __builtin_amdgcn_wmma_f32_16x16x32_bf16(false, A[c].v, false, Bm[c].v,
                                                  (short)0, acc, false, false);
  const float bv = bias[node * DD + e];
  // D layout: VGPR r -> row M = r + 8*half, col = e
  unsigned short* dp = dstSlot + half * 8 * STRIDE + e;
#pragma unroll
  for (int r = 0; r < 8; r += 2) {
    float x0 = __builtin_fmaxf(acc[r] + bv, 0.f);
    float x1 = __builtin_fmaxf(acc[r + 1] + bv, 0.f);
    unsigned pr = pack2(x0, x1);
    dp[r * STRIDE]       = (unsigned short)(pr & 0xffffu);
    dp[(r + 1) * STRIDE] = (unsigned short)(pr >> 16);
  }
}

// prologue: W_all f32 -> bf16 in workspace (reused by all 256 row-tiles via L2)
__global__ __launch_bounds__(256) void wcvt_kernel(const float* __restrict__ W,
                                                   unsigned short* __restrict__ Wb) {
  const size_t i = ((size_t)blockIdx.x * 256 + threadIdx.x) * 8;
  float4 a = *(const float4*)(W + i);
  float4 b = *(const float4*)(W + i + 4);
  uint4 o;
  o.x = pack2(a.x, a.y); o.y = pack2(a.z, a.w);
  o.z = pack2(b.x, b.y); o.w = pack2(b.z, b.w);
  *(uint4*)(Wb + i) = o;
}

// main: one workgroup owns 16 batch rows and runs the whole 65-node DAG in LDS
__global__ __launch_bounds__(256) void dag_kernel(const float* __restrict__ X,
                                                  const unsigned short* __restrict__ Wb,
                                                  const float* __restrict__ bias,
                                                  float* __restrict__ out) {
  extern __shared__ unsigned short smem[];
  unsigned short* nodeOut = smem;                  // [NN][ROWS][STRIDE] bf16
  unsigned short* stage   = smem + NODEOUT_ELEMS;  // [K_NODES][ROWS][STRIDE] bf16

  const int tid  = threadIdx.x;
  const int wave = tid >> 5;     // 8 waves (wave32)
  const int lane = tid & 31;
  const int rowBase = blockIdx.x * ROWS;
  const int sm = tid >> 4;       // staging row 0..15
  const int sd = (tid & 15) * 8; // staging col start
  const int sOff = sm * STRIDE + sd;

  // ---- node 0 input: bf16(X tile) -> stage slot 0 (coalesced f32 loads)
  {
    const float* xp = X + (size_t)(rowBase + sm) * DD + sd;
    float4 a = *(const float4*)xp;
    float4 b = *(const float4*)(xp + 4);
    uint4 o;
    o.x = pack2(a.x, a.y); o.y = pack2(a.z, a.w);
    o.z = pack2(b.x, b.y); o.w = pack2(b.z, b.w);
    *(uint4*)(stage + sOff) = o;
  }
  __syncthreads();
  { // node 0: wave w computes column tile w
    ABReg A[4];
    load_A(stage, lane, A);
    mm_tile(A, 0, wave, lane, Wb, bias, nodeOut);
  }
  __syncthreads();

  int nodeBase = 1;
  for (int l = 0; l < L_LAYERS; l++) {
    // warm L2/WGP$ with this layer's weight panel (global_prefetch_b8)
    {
      const unsigned short* wpan = Wb + (size_t)nodeBase * DD * DD;
#pragma unroll
      for (int pf = 0; pf < 4; pf++)
        __builtin_prefetch(wpan + ((size_t)tid * 4 + pf) * 128, 0, 3);
    }
    // stage V_in for all 8 nodes of this layer: packed-bf16 pairwise parent sums
    for (int k = 0; k < K_NODES; k++) {
      uint4 q0 = *(const uint4*)(nodeOut + g_srcs.v[l][k][0] * SLICE + sOff);
      uint4 q1 = *(const uint4*)(nodeOut + g_srcs.v[l][k][1] * SLICE + sOff);
      uint4 q2 = *(const uint4*)(nodeOut + g_srcs.v[l][k][2] * SLICE + sOff);
      uint4 q3 = *(const uint4*)(nodeOut + g_srcs.v[l][k][3] * SLICE + sOff);
      uint4 o;
      o.x = pk_add_bf16(pk_add_bf16(q0.x, q1.x), pk_add_bf16(q2.x, q3.x));
      o.y = pk_add_bf16(pk_add_bf16(q0.y, q1.y), pk_add_bf16(q2.y, q3.y));
      o.z = pk_add_bf16(pk_add_bf16(q0.z, q1.z), pk_add_bf16(q2.z, q3.z));
      o.w = pk_add_bf16(pk_add_bf16(q0.w, q1.w), pk_add_bf16(q2.w, q3.w));
      *(uint4*)(stage + k * SLICE + sOff) = o;
    }
    __syncthreads();
    { // wave w owns node (nodeBase+w): A shared across its 8 column tiles
      const int node = nodeBase + wave;
      ABReg A[4];
      load_A(stage + wave * SLICE, lane, A);
      unsigned short* dst = nodeOut + node * SLICE;
#pragma unroll
      for (int t = 0; t < K_NODES; t++)
        mm_tile(A, node, t, lane, Wb, bias, dst);
    }
    __syncthreads();
    nodeBase += K_NODES;
  }

  // mean of last 8 node outputs -> f32 d_out (vector LDS loads, f32 accumulate)
  {
    float s[8] = {0.f, 0.f, 0.f, 0.f, 0.f, 0.f, 0.f, 0.f};
#pragma unroll
    for (int j = 0; j < K_NODES; j++) {
      uint4 q = *(const uint4*)(nodeOut + (NN - K_NODES + j) * SLICE + sOff);
      s[0] += bflo(q.x); s[1] += bfhi(q.x);
      s[2] += bflo(q.y); s[3] += bfhi(q.y);
      s[4] += bflo(q.z); s[5] += bfhi(q.z);
      s[6] += bflo(q.w); s[7] += bfhi(q.w);
    }
    float* op = out + (size_t)(rowBase + sm) * DD + sd;
    float4 o0 = {s[0] * 0.125f, s[1] * 0.125f, s[2] * 0.125f, s[3] * 0.125f};
    float4 o1 = {s[4] * 0.125f, s[5] * 0.125f, s[6] * 0.125f, s[7] * 0.125f};
    *(float4*)op = o0;
    *(float4*)(op + 4) = o1;
  }
}

extern "C" void kernel_launch(void* const* d_in, const int* in_sizes, int n_in,
                              void* d_out, int out_size, void* d_ws, size_t ws_size,
                              hipStream_t stream) {
  (void)in_sizes; (void)n_in; (void)out_size; (void)ws_size;
  const float* X    = (const float*)d_in[0];
  const float* W    = (const float*)d_in[1];
  const float* bias = (const float*)d_in[2];
  float* out = (float*)d_out;
  unsigned short* Wb = (unsigned short*)d_ws;  // needs NN*DD*DD*2 = ~2.1 MB

  wcvt_kernel<<<dim3((NN * DD * DD) / (256 * 8)), dim3(256), 0, stream>>>(W, Wb);
  dag_kernel<<<dim3(BB / ROWS), dim3(256), SMEM_BYTES, stream>>>(X, Wb, bias, out);
}
